// SparseAttention_36885179138501
// MI455X (gfx1250) — compile-verified, matrix operand order checked by reading, not run on hardware
//
#include <hip/hip_runtime.h>

// ---------------------------------------------------------------------------
// Problem constants (reference: B=2, S=4096, D=512, H=8, WIN=50, DK=64)
// ---------------------------------------------------------------------------
constexpr int Bc  = 2;
constexpr int Sc  = 4096;
constexpr int Dc  = 512;
constexpr int Hc  = 8;
constexpr int DKc = 64;
constexpr int WINc = 50;
constexpr int Mc  = Bc * Sc;          // 8192 rows for the projection GEMMs

typedef __attribute__((ext_vector_type(16))) __bf16    v16bf;
typedef __attribute__((ext_vector_type(8)))  float     v8f;
typedef __attribute__((ext_vector_type(4)))  unsigned  vuint4;

union TileBits { vuint4 q[2]; v16bf bf; };

// fp32 -> bf16 round-to-nearest-even
static __device__ inline unsigned short f2bf(float f) {
  unsigned int u = __float_as_uint(f);
  unsigned int r = u + 0x7fffu + ((u >> 16) & 1u);
  return (unsigned short)(r >> 16);
}

// Load a 16x32 bf16 tile in the CDNA5 WMMA A/B per-lane layout:
// lane L -> row (L & 15); K halves: (L<16 ? k0..7,k16..23 : k8..15,k24..31)
// Two 16-byte loads per lane (global_load_b128).
static __device__ inline v16bf load_tile16x32(const unsigned short* __restrict__ base,
                                              int stride) {
  const int lane = threadIdx.x & 31;
  const unsigned short* p = base + (lane & 15) * stride + ((lane >> 4) << 3);
  TileBits t;
  t.q[0] = *reinterpret_cast<const vuint4*>(p);
  t.q[1] = *reinterpret_cast<const vuint4*>(p + 16);
  return t.bf;
}

static __device__ inline v8f wmma_bf16(v16bf a, v16bf b, v8f c) {
  return __builtin_amdgcn_wmma_f32_16x16x32_bf16(
      /*neg_a=*/false, a, /*neg_b=*/false, b,
      /*c_mod=*/(short)0, c, /*reuse_a=*/false, /*reuse_b=*/false);
}

// ---------------------------------------------------------------------------
// Elementwise fp32 -> bf16 conversion
// ---------------------------------------------------------------------------
__global__ void cvt_f32_bf16(const float* __restrict__ in,
                             unsigned short* __restrict__ out, int n) {
  int i = blockIdx.x * 256 + threadIdx.x;
  if (i < n) out[i] = f2bf(in[i]);
}

// ---------------------------------------------------------------------------
// ctrl[j] = any_b (opcode[b, j] == 0)
// ---------------------------------------------------------------------------
__global__ void build_ctrl(const long long* __restrict__ opc,
                           unsigned char* __restrict__ ctrl) {
  int j = blockIdx.x * 256 + threadIdx.x;
  if (j < Sc) {
    unsigned char c = 0;
    for (int b = 0; b < Bc; ++b) c |= (opc[(size_t)b * Sc + j] == 0);
    ctrl[j] = c;
  }
}

// ---------------------------------------------------------------------------
// Epilogue: store one 16x16 f32 C tile per mode.
//   mode 0: Q  -> bf16 [M,D], scaled by 1/sqrt(DK)
//   mode 1: K  -> bf16 [M,D]
//   mode 2: V  -> bf16 transposed [B,H,DK,S]
//   mode 3: O  -> fp32 [M,D] (final output)
// ---------------------------------------------------------------------------
static __device__ inline void store_ctile(int mode, void* __restrict__ out,
                                          const float* __restrict__ bias,
                                          int m0, int n0, v8f acc) {
  const int lane  = threadIdx.x & 31;
  const int col   = n0 + (lane & 15);
  const float bv  = bias[col];
  const int rbase = m0 + ((lane >> 4) << 3);

  if (mode == 3) {
    float* o = (float*)out;
#pragma unroll
    for (int e = 0; e < 8; ++e)
      o[(size_t)(rbase + e) * Dc + col] = acc[e] + bv;
  } else if (mode == 2) {
    unsigned short* o = (unsigned short*)out;
    const int h = col / DKc, dk = col % DKc;
#pragma unroll
    for (int e = 0; e < 8; ++e) {
      int m = rbase + e;
      int b = m / Sc, s = m % Sc;
      o[((size_t)((b * Hc + h) * DKc + dk)) * Sc + s] = f2bf(acc[e] + bv);
    }
  } else {
    unsigned short* o = (unsigned short*)out;
    const float scale = (mode == 0) ? 0.125f : 1.0f;   // 1/sqrt(DK) folded into Q
#pragma unroll
    for (int e = 0; e < 8; ++e)
      o[(size_t)(rbase + e) * Dc + col] = f2bf((acc[e] + bv) * scale);
  }
}

// ---------------------------------------------------------------------------
// out = A(bf16, MxK) @ W(bf16, NxK)^T + bias
// Each wave computes a 32x64 output strip (2 M-tiles x 4 N-tiles):
// per k-step 6 tile loads (12 b128) feed 8 WMMAs -> 1.5 b128/WMMA,
// 8 f32 accumulators (64 VGPRs) held in registers.
// ---------------------------------------------------------------------------
__global__ void __launch_bounds__(256)
gemm_xwt(const unsigned short* __restrict__ A,
         const unsigned short* __restrict__ W,
         const float* __restrict__ bias,
         void* __restrict__ out, int mode) {
  const int wave = threadIdx.x >> 5;
  const int tile = blockIdx.x * 8 + wave;
  const int ntn  = Dc / 64;                    // 8 N-strips
  const int m0   = (tile / ntn) * 32;
  const int n0   = (tile % ntn) * 64;

  v8f acc[2][4] = {};
  for (int k0 = 0; k0 < Dc; k0 += 32) {
    v16bf a0 = load_tile16x32(A + (size_t)m0 * Dc + k0, Dc);
    v16bf a1 = load_tile16x32(A + (size_t)(m0 + 16) * Dc + k0, Dc);
#pragma unroll
    for (int j = 0; j < 4; ++j) {
      v16bf b = load_tile16x32(W + (size_t)(n0 + 16 * j) * Dc + k0, Dc);
      acc[0][j] = wmma_bf16(a0, b, acc[0][j]);
      acc[1][j] = wmma_bf16(a1, b, acc[1][j]);
    }
  }

#pragma unroll
  for (int i = 0; i < 2; ++i)
#pragma unroll
    for (int j = 0; j < 4; ++j)
      store_ctile(mode, out, bias, m0 + 16 * i, n0 + 16 * j, acc[i][j]);
}

// ---------------------------------------------------------------------------
// Flash attention with element-wise sparse mask.
// One wave per (b, h, 16-query tile); key loop in steps of 32.
//   scores: Q(16x64) . K(32x64)^T  -> two 16x16 WMMAs per key sub-tile
//   P.V   : P(16x32 bf16 via LDS transpose) . V(32x64) -> 4 WMMAs
// ---------------------------------------------------------------------------
constexpr int AW = 4;   // waves per block

__global__ void __launch_bounds__(AW * 32)
attn_sparse(const unsigned short* __restrict__ Q,
            const unsigned short* __restrict__ Kb,
            const unsigned short* __restrict__ Vt,
            const unsigned char* __restrict__ ctrl,
            unsigned short* __restrict__ Ob) {
  __shared__ unsigned short lds_p[AW][16 * 32];

  const int lane = threadIdx.x & 31;
  const int wave = threadIdx.x >> 5;
  const int task = blockIdx.x * AW + wave;
  const int qtiles = Sc / 16;                 // 256
  const int bh = task / qtiles;
  const int i0 = (task % qtiles) * 16;
  const int b  = bh / Hc, h = bh % Hc;

  // Q operand tiles (already scaled by 1/sqrt(DK))
  const unsigned short* qbase = Q + (size_t)(b * Sc + i0) * Dc + h * DKc;
  const v16bf aq0 = load_tile16x32(qbase, Dc);
  const v16bf aq1 = load_tile16x32(qbase + 32, Dc);

  v8f accO[4] = {};
  float mrow[8], lrow[8];
#pragma unroll
  for (int e = 0; e < 8; ++e) { mrow[e] = -3.0e38f; lrow[e] = 0.0f; }

  const unsigned short* kbase = Kb + (size_t)(b * Sc) * Dc + h * DKc;
  const unsigned short* vbase = Vt + (size_t)(bh * DKc) * Sc;

  const int ib = i0 + ((lane >> 4) << 3);     // base query row for this lane

  for (int j0 = 0; j0 < Sc; j0 += 32) {
    // ---- tile skip: window overlap OR any ctrl column in tile ----
    const int clv = (int)ctrl[j0 + lane];                 // 32 lanes cover tile
    const bool window_hit = (j0 <= i0 + 15 + WINc) && (j0 + 31 >= i0 - WINc);
    const bool any_ctrl = (__ballot(clv != 0) != 0ull);
    if (!window_hit && !any_ctrl) continue;

    const int c0f = __shfl(clv, lane & 15);               // ctrl for col j0+(lane&15)
    const int c1f = __shfl(clv, (lane & 15) + 16);        // ctrl for col j0+16+(lane&15)

    // ---- scores: two 16x16 tiles covering 32 keys ----
    v8f s0 = {}, s1 = {};
    {
      v16bf bk = load_tile16x32(kbase + (size_t)j0 * Dc, Dc);
      s0 = wmma_bf16(aq0, bk, s0);
      bk = load_tile16x32(kbase + (size_t)j0 * Dc + 32, Dc);
      s0 = wmma_bf16(aq1, bk, s0);
      bk = load_tile16x32(kbase + (size_t)(j0 + 16) * Dc, Dc);
      s1 = wmma_bf16(aq0, bk, s1);
      bk = load_tile16x32(kbase + (size_t)(j0 + 16) * Dc + 32, Dc);
      s1 = wmma_bf16(aq1, bk, s1);
    }

    // ---- mask + online softmax stats ----
    const int jc0 = j0 + (lane & 15), jc1 = jc0 + 16;
    float p0[8], p1[8], newm[8], scl[8];
#pragma unroll
    for (int e = 0; e < 8; ++e) {
      const int i  = ib + e;
      const int d0 = i - jc0, d1 = i - jc1;
      const bool a0 = (d0 <= WINc && d0 >= -WINc) || (c0f != 0);
      const bool a1 = (d1 <= WINc && d1 >= -WINc) || (c1f != 0);
      p0[e] = a0 ? s0[e] : -1e9f;
      p1[e] = a1 ? s1[e] : -1e9f;
    }
#pragma unroll
    for (int e = 0; e < 8; ++e) {             // row max across 16 lanes
      float mx = fmaxf(p0[e], p1[e]);
      mx = fmaxf(mx, __shfl_xor(mx, 1));
      mx = fmaxf(mx, __shfl_xor(mx, 2));
      mx = fmaxf(mx, __shfl_xor(mx, 4));
      mx = fmaxf(mx, __shfl_xor(mx, 8));
      const float mn = fmaxf(mrow[e], mx);
      newm[e] = mn;
      scl[e]  = __expf(mrow[e] - mn);
    }
#pragma unroll
    for (int e = 0; e < 8; ++e) {             // p = exp(s - m), row sums, l update
      const float e0 = __expf(p0[e] - newm[e]);
      const float e1 = __expf(p1[e] - newm[e]);
      p0[e] = e0; p1[e] = e1;
      float rs = e0 + e1;
      rs += __shfl_xor(rs, 1);
      rs += __shfl_xor(rs, 2);
      rs += __shfl_xor(rs, 4);
      rs += __shfl_xor(rs, 8);
      lrow[e] = lrow[e] * scl[e] + rs;
      mrow[e] = newm[e];
    }
#pragma unroll
    for (int t = 0; t < 4; ++t)
#pragma unroll
      for (int e = 0; e < 8; ++e) accO[t][e] *= scl[e];

    // ---- P: C-layout -> bf16 A-layout via wave-private LDS bounce ----
    unsigned short* lp = lds_p[wave];
#pragma unroll
    for (int e = 0; e < 8; ++e) {
      const int r = ((lane >> 4) << 3) + e;
      lp[r * 32 + (lane & 15)]      = f2bf(p0[e]);
      lp[r * 32 + 16 + (lane & 15)] = f2bf(p1[e]);
    }
    asm volatile("s_wait_dscnt 0" ::: "memory");   // wave-local DS RAW fence
    const v16bf ap = load_tile16x32(lp, 32);

    // ---- accO += P . V (V pre-transposed: rows = dk, cols = s) ----
#pragma unroll
    for (int t = 0; t < 4; ++t) {
      v16bf bvt = load_tile16x32(vbase + (size_t)(t * 16) * Sc + j0, Sc);
      accO[t] = wmma_bf16(ap, bvt, accO[t]);
    }
  }

  // ---- normalize and store attention output (bf16 [M, D]) ----
  float inv_l[8];
#pragma unroll
  for (int e = 0; e < 8; ++e) inv_l[e] = 1.0f / lrow[e];   // diagonal always allowed

  unsigned short* ob = Ob + (size_t)(b * Sc + i0) * Dc + h * DKc;
#pragma unroll
  for (int t = 0; t < 4; ++t) {
    const int dk = t * 16 + (lane & 15);
#pragma unroll
    for (int e = 0; e < 8; ++e) {
      const int r = ((lane >> 4) << 3) + e;
      ob[(size_t)r * Dc + dk] = f2bf(accO[t][e] * inv_l[e]);
    }
  }
}

// ---------------------------------------------------------------------------
// Host launcher
// ---------------------------------------------------------------------------
extern "C" void kernel_launch(void* const* d_in, const int* in_sizes, int n_in,
                              void* d_out, int out_size, void* d_ws, size_t ws_size,
                              hipStream_t stream) {
  const float*     x   = (const float*)d_in[0];
  const long long* opc = (const long long*)d_in[1];
  const float*     Wq  = (const float*)d_in[2];
  const float*     bq  = (const float*)d_in[3];
  const float*     Wk  = (const float*)d_in[4];
  const float*     bk  = (const float*)d_in[5];
  const float*     Wv  = (const float*)d_in[6];
  const float*     bv  = (const float*)d_in[7];
  const float*     Wo  = (const float*)d_in[8];
  const float*     bo  = (const float*)d_in[9];

  const size_t MK = (size_t)Mc * Dc;       // 4,194,304 elements
  const size_t WW = (size_t)Dc * Dc;       //   262,144 elements

  char* w = (char*)d_ws;
  unsigned short* xb  = (unsigned short*)w; w += MK * 2;   // x  bf16
  unsigned short* wqb = (unsigned short*)w; w += WW * 2;   // Wq bf16
  unsigned short* wkb = (unsigned short*)w; w += WW * 2;
  unsigned short* wvb = (unsigned short*)w; w += WW * 2;
  unsigned short* wob = (unsigned short*)w; w += WW * 2;
  unsigned short* qb  = (unsigned short*)w; w += MK * 2;   // Q bf16 (pre-scaled)
  unsigned short* kb  = (unsigned short*)w; w += MK * 2;   // K bf16
  unsigned short* vt  = (unsigned short*)w; w += MK * 2;   // V bf16 [B,H,DK,S]
  unsigned short* ob  = (unsigned short*)w; w += MK * 2;   // attn out bf16
  unsigned char*  ctl = (unsigned char*)w; w += Sc;

  // 1) conversions
  cvt_f32_bf16<<<(int)((MK + 255) / 256), 256, 0, stream>>>(x,  xb,  (int)MK);
  cvt_f32_bf16<<<(int)((WW + 255) / 256), 256, 0, stream>>>(Wq, wqb, (int)WW);
  cvt_f32_bf16<<<(int)((WW + 255) / 256), 256, 0, stream>>>(Wk, wkb, (int)WW);
  cvt_f32_bf16<<<(int)((WW + 255) / 256), 256, 0, stream>>>(Wv, wvb, (int)WW);
  cvt_f32_bf16<<<(int)((WW + 255) / 256), 256, 0, stream>>>(Wo, wob, (int)WW);

  // 2) sparse-mask column flags
  build_ctrl<<<(Sc + 255) / 256, 256, 0, stream>>>(opc, ctl);

  // 3) projections (WMMA): strips = (M/32)*(D/64) = 2048, 8 waves/block
  const int gemm_blocks = (Mc / 32) * (Dc / 64) / 8;       // 256
  gemm_xwt<<<gemm_blocks, 256, 0, stream>>>(xb, wqb, bq, qb, 0);
  gemm_xwt<<<gemm_blocks, 256, 0, stream>>>(xb, wkb, bk, kb, 1);
  gemm_xwt<<<gemm_blocks, 256, 0, stream>>>(xb, wvb, bv, vt, 2);

  // 4) flash attention: B*H*(S/16) = 4096 wave-tasks, 4 waves/block
  attn_sparse<<<(Bc * Hc * (Sc / 16)) / AW, AW * 32, 0, stream>>>(qb, kb, vt, ctl, ob);

  // 5) output projection to fp32 d_out
  gemm_xwt<<<gemm_blocks, 256, 0, stream>>>(ob, wob, bo, (float*)d_out, 3);
}
